// VoiceModel_31542239821860
// MI455X (gfx1250) — matrix operand order repaired and checked by reading, not compiled, not run on hardware
//
#include <hip/hip_runtime.h>
#include <hip/hip_bf16.h>

#define B_   256
#define V_   4
#define T_   512
#define N_   128   // class axis = notes-LSTM sequence length
#define H_   256
#define HL_  200

typedef __attribute__((ext_vector_type(16))) __bf16 v16bf;
typedef __attribute__((ext_vector_type(8)))  __bf16 v8bf;
typedef __attribute__((ext_vector_type(8)))  float  v8f;

// explicit global-address-space pointer types (avoid flat_load/flat_store)
typedef const __attribute__((address_space(1))) v8bf*  gv8bf_cp;
typedef const __attribute__((address_space(1))) v16bf* gv16bf_cp;
typedef       __attribute__((address_space(1))) __bf16* gbf16_p;

__device__ __forceinline__ float sigf(float x) { return 1.0f / (1.0f + __expf(-x)); }

// ---------------------------------------------------------------------------
// Workspace zero-init (h double-buffers + sync counters). Runs every launch.
// ---------------------------------------------------------------------------
__global__ void init_ws_kernel(__bf16* hbufs, unsigned* ctr) {
  unsigned i = blockIdx.x * blockDim.x + threadIdx.x;
  if (i < 4u * 2u * B_ * H_) hbufs[i] = (__bf16)0.0f;
  if (i < 16) ctr[i] = 0u;
}

// ---------------------------------------------------------------------------
// Scalar 2-layer LSTM (hidden size 1) over one-hot series.
// One block = one (batch, branch); 128 threads = one class each.
// ---------------------------------------------------------------------------
__global__ void __launch_bounds__(128) scalar_lstm_kernel(
    const int* __restrict__ ln, const int* __restrict__ rn,
    const float* __restrict__ tlWih, const float* __restrict__ tlWhh, const float* __restrict__ tlB,
    const float* __restrict__ trWih, const float* __restrict__ trWhh, const float* __restrict__ trB,
    float* __restrict__ states)   // [2][B][N]
{
  __shared__ int toks[T_];
  const int b = blockIdx.x, br = blockIdx.y, c = threadIdx.x;
  const int* tok = (br == 0 ? ln : rn) + (size_t)b * V_ * T_;   // voice 0 row
  for (int i = threadIdx.x; i < T_; i += blockDim.x) toks[i] = tok[i];
  __syncthreads();

  const float* Wih = br == 0 ? tlWih : trWih;
  const float* Whh = br == 0 ? tlWhh : trWhh;
  const float* Bb  = br == 0 ? tlB   : trB;
  float wi0[4], wh0[4], bb0[4], wi1[4], wh1[4], bb1[4];
#pragma unroll
  for (int g = 0; g < 4; ++g) {
    wi0[g] = Wih[g];     wh0[g] = Whh[g];     bb0[g] = Bb[g];
    wi1[g] = Wih[4 + g]; wh1[g] = Whh[4 + g]; bb1[g] = Bb[4 + g];
  }
  float h0 = 0.f, c0 = 0.f, h1 = 0.f, c1 = 0.f;
  for (int t = 0; t < T_; ++t) {
    const float x = (toks[t] == c) ? 1.0f : 0.0f;
    float gi = sigf(wi0[0] * x + wh0[0] * h0 + bb0[0]);
    float gf = sigf(wi0[1] * x + wh0[1] * h0 + bb0[1]);
    float gg = tanhf(wi0[2] * x + wh0[2] * h0 + bb0[2]);
    float go = sigf(wi0[3] * x + wh0[3] * h0 + bb0[3]);
    c0 = gf * c0 + gi * gg;
    h0 = go * tanhf(c0);
    gi = sigf(wi1[0] * h0 + wh1[0] * h1 + bb1[0]);
    gf = sigf(wi1[1] * h0 + wh1[1] * h1 + bb1[1]);
    gg = tanhf(wi1[2] * h0 + wh1[2] * h1 + bb1[2]);
    go = sigf(wi1[3] * h0 + wh1[3] * h1 + bb1[3]);
    c1 = gf * c1 + gi * gg;
    h1 = go * tanhf(c1);
  }
  states[((size_t)br * B_ + b) * N_ + c] = h1;
}

// ---------------------------------------------------------------------------
// Pack one (4H, H) fp32 weight matrix into bf16 WMMA B-fragment order.
// Block (gate g, 16-col tile ct, K-chunk kc) = 512 bf16; lane L supplies cols
// ct*16+(L&15), K = kc*32 + (L>=16?8:0) + (e&7) + (e>=8?16:0) (e=0..15) per ISA B 32x16 striping.
// ---------------------------------------------------------------------------
__global__ void pack_weight_kernel(const float* __restrict__ W, __bf16* __restrict__ dst) {
  const int idx = blockIdx.x * blockDim.x + threadIdx.x;   // 0 .. 262143
  const int e    = idx & 15;
  const int lane = (idx >> 4) & 31;
  const int blk  = idx >> 9;        // 0..511
  const int kc   = blk & 7;
  const int gct  = blk >> 3;        // g*16 + ct
  const int g    = gct >> 4, ct = gct & 15;
  const int row  = g * H_ + ct * 16 + (lane & 15);          // gate-output row of W
  const int col  = kc * 32 + ((lane >> 4) << 3) + (e & 7) + ((e >> 3) << 4);   // hidden (K) index
  dst[idx] = (__bf16)W[(size_t)row * H_ + col];
}

// ---------------------------------------------------------------------------
// Persistent 2-layer notes-LSTM with WMMA.
// ---------------------------------------------------------------------------
__device__ __forceinline__ v16bf load_a_frag(const __bf16* h, int row, int kc, int koff) {
  gv8bf_cp gp = (gv8bf_cp)(unsigned long long)(h + (size_t)row * H_ + kc * 32 + koff);
  const v8bf lo = gp[0];
  const v8bf hi = gp[2];   // +32 bytes = +16 elements
  v16bf a;
#pragma unroll
  for (int e = 0; e < 8; ++e) { a[e] = lo[e]; a[e + 8] = hi[e]; }
  return a;
}

__device__ __forceinline__ v16bf load_b_frag(const __bf16* W, int g, int ct, int kc, int lane) {
  const __bf16* p = W + ((size_t)((g * 16 + ct) * 8 + kc)) * 512 + lane * 16;
  return *(gv16bf_cp)(unsigned long long)p;
}

__device__ __forceinline__ void store_h_tile(__bf16* h, int rb, int roff, int hj, const v8f& v) {
#pragma unroll
  for (int r = 0; r < 8; ++r) {
    gbf16_p p = (gbf16_p)(unsigned long long)(h + (size_t)(rb + r + roff) * H_ + hj);
    *p = (__bf16)v[r];
  }
}

__device__ __forceinline__ void group_barrier(unsigned* bar, unsigned target) {
  __builtin_amdgcn_fence(__ATOMIC_RELEASE, "agent");
  __syncthreads();
  if (threadIdx.x == 0) {
    atomicAdd(bar, 1u);
    while (((volatile unsigned*)bar)[0] < target) __builtin_amdgcn_s_sleep(1);
  }
  __syncthreads();
  __builtin_amdgcn_fence(__ATOMIC_ACQUIRE, "agent");
}

struct WaveCtx {
  const float*  st;
  const __bf16* W0;
  const __bf16* W1;
  const __bf16* W2;
  unsigned*     bar;
  int rb, arow, hj, ct, koff, roff, lane;
  float wih0g[4], b0g[4], b1g[4];
};

__device__ __forceinline__ void lstm_step(const WaveCtx& cx, int t, unsigned& phase,
                                          v8f& c0, v8f& c1,
                                          const __bf16* h0r, __bf16* h0w,
                                          const __bf16* h1r, __bf16* h1w)
{
  // ---- layer 0: acc = x_t * Wih0 + b0 (outer product folded into C) ----
  v8f acc[4];
#pragma unroll
  for (int v = 0; v < 8; ++v) {
    const float s = cx.st[(size_t)(cx.rb + v + cx.roff) * N_ + t];
#pragma unroll
    for (int g = 0; g < 4; ++g) acc[g][v] = s * cx.wih0g[g] + cx.b0g[g];
  }
#pragma unroll 2
  for (int kc = 0; kc < 8; ++kc) {
    __builtin_prefetch(cx.W0 + ((size_t)(cx.ct * 8) + kc + 1) * 512 + cx.lane * 16, 0, 0);
    const v16bf a = load_a_frag(h0r, cx.arow, kc, cx.koff);
#pragma unroll
    for (int g = 0; g < 4; ++g)
      acc[g] = __builtin_amdgcn_wmma_f32_16x16x32_bf16(false, a, false,
                 load_b_frag(cx.W0, g, cx.ct, kc, cx.lane), (short)0, acc[g], false, false);
  }
  v8f hv;
#pragma unroll
  for (int v = 0; v < 8; ++v) {
    const float gi = sigf(acc[0][v]), gf = sigf(acc[1][v]);
    const float gg = tanhf(acc[2][v]), go = sigf(acc[3][v]);
    const float cn = gf * c0[v] + gi * gg;
    c0[v] = cn;
    hv[v] = go * tanhf(cn);
  }
  store_h_tile(h0w, cx.rb, cx.roff, cx.hj, hv);
  group_barrier(cx.bar, ++phase * 8u);

  // ---- layer 1: acc = b1 + h0_new @ Wih1^T + h1_old @ Whh1^T ----
#pragma unroll
  for (int g = 0; g < 4; ++g)
#pragma unroll
    for (int v = 0; v < 8; ++v) acc[g][v] = cx.b1g[g];
#pragma unroll 2
  for (int kc = 0; kc < 8; ++kc) {
    __builtin_prefetch(cx.W1 + ((size_t)(cx.ct * 8) + kc + 1) * 512 + cx.lane * 16, 0, 0);
    const v16bf a = load_a_frag(h0w, cx.arow, kc, cx.koff);   // h0 new
#pragma unroll
    for (int g = 0; g < 4; ++g)
      acc[g] = __builtin_amdgcn_wmma_f32_16x16x32_bf16(false, a, false,
                 load_b_frag(cx.W1, g, cx.ct, kc, cx.lane), (short)0, acc[g], false, false);
  }
#pragma unroll 2
  for (int kc = 0; kc < 8; ++kc) {
    __builtin_prefetch(cx.W2 + ((size_t)(cx.ct * 8) + kc + 1) * 512 + cx.lane * 16, 0, 0);
    const v16bf a = load_a_frag(h1r, cx.arow, kc, cx.koff);   // h1 old
#pragma unroll
    for (int g = 0; g < 4; ++g)
      acc[g] = __builtin_amdgcn_wmma_f32_16x16x32_bf16(false, a, false,
                 load_b_frag(cx.W2, g, cx.ct, kc, cx.lane), (short)0, acc[g], false, false);
  }
#pragma unroll
  for (int v = 0; v < 8; ++v) {
    const float gi = sigf(acc[0][v]), gf = sigf(acc[1][v]);
    const float gg = tanhf(acc[2][v]), go = sigf(acc[3][v]);
    const float cn = gf * c1[v] + gi * gg;
    c1[v] = cn;
    hv[v] = go * tanhf(cn);
  }
  store_h_tile(h1w, cx.rb, cx.roff, cx.hj, hv);
  group_barrier(cx.bar, ++phase * 8u);
}

__global__ void __launch_bounds__(256) notes_lstm_kernel(
    const float* __restrict__ states,       // [2][B][N]
    const __bf16* __restrict__ packW,       // [2 branches][3 mats][262144]
    const float* __restrict__ wih0L, const float* __restrict__ b0L, const float* __restrict__ b1L,
    const float* __restrict__ wih0R, const float* __restrict__ b0R, const float* __restrict__ b1R,
    __bf16* __restrict__ h0p0, __bf16* __restrict__ h0p1,
    __bf16* __restrict__ h1p0, __bf16* __restrict__ h1p1,
    unsigned* __restrict__ ctr)
{
  const int idx  = blockIdx.x;          // 0..63
  const int br   = idx >> 5;            // branch
  const int rem  = idx & 31;
  const int bt   = rem >> 3;            // batch tile (64 rows)
  const int hs   = rem & 7;             // hidden slice (32 cols)
  const int lane = threadIdx.x & 31;
  const int w    = threadIdx.x >> 5;    // wave 0..7

  WaveCtx cx;
  cx.rb   = bt * 64 + (w >> 1) * 16;
  cx.ct   = hs * 2 + (w & 1);
  cx.arow = cx.rb + (lane & 15);
  cx.hj   = cx.ct * 16 + (lane & 15);
  cx.koff = (lane >= 16) ? 8 : 0;
  cx.roff = (lane >= 16) ? 8 : 0;
  cx.lane = lane;
  cx.W0   = packW + ((size_t)br * 3 + 0) * 262144;  // Whh0
  cx.W1   = packW + ((size_t)br * 3 + 1) * 262144;  // Wih1
  cx.W2   = packW + ((size_t)br * 3 + 2) * 262144;  // Whh1
  cx.st   = states + (size_t)br * B_ * N_;
  cx.bar  = ctr + (br * 4 + bt);

  const float* wih0 = br ? wih0R : wih0L;
  const float* b0   = br ? b0R   : b0L;
  const float* b1   = br ? b1R   : b1L;
#pragma unroll
  for (int g = 0; g < 4; ++g) {
    const int gc = g * H_ + cx.hj;
    cx.wih0g[g] = wih0[gc]; cx.b0g[g] = b0[gc]; cx.b1g[g] = b1[gc];
  }

  const size_t hoff = (size_t)br * (B_ * H_);
  __bf16* h0A = h0p0 + hoff;  __bf16* h0B = h0p1 + hoff;
  __bf16* h1A = h1p0 + hoff;  __bf16* h1B = h1p1 + hoff;

  unsigned phase = 0;
  v8f c0 = {}; v8f c1 = {};   // cell states live in accumulator layout

  for (int t = 0; t < N_; t += 2) {
    lstm_step(cx, t,     phase, c0, c1, h0A, h0B, h1A, h1B);
    lstm_step(cx, t + 1, phase, c0, c1, h0B, h0A, h1B, h1A);
  }
  // final h1 ends up in h1A == h1p0 (+hoff)
}

// ---------------------------------------------------------------------------
// MLP head: out = relu([h_ln, h_rn] @ W1^T + b1) @ W2^T + b2
// ---------------------------------------------------------------------------
__global__ void __launch_bounds__(256) mlp_kernel(
    const __bf16* __restrict__ h1final,   // [2][B][H] (parity-0 buffer)
    const float* __restrict__ W1, const float* __restrict__ bb1,
    const float* __restrict__ W2, const float* __restrict__ bb2,
    float* __restrict__ out)
{
  __shared__ float x[2 * H_];
  __shared__ float hid[HL_];
  const int b = blockIdx.x, tid = threadIdx.x;
  x[tid]      = (float)h1final[(size_t)b * H_ + tid];
  x[H_ + tid] = (float)h1final[(size_t)(B_ + b) * H_ + tid];
  __syncthreads();
  if (tid < HL_) {
    float a = bb1[tid];
    const float* wr = W1 + (size_t)tid * 2 * H_;
    for (int k = 0; k < 2 * H_; ++k) a += x[k] * wr[k];
    hid[tid] = fmaxf(a, 0.0f);
  }
  __syncthreads();
  if (tid < N_) {
    float a = bb2[tid];
    const float* wr = W2 + (size_t)tid * HL_;
    for (int k = 0; k < HL_; ++k) a += hid[k] * wr[k];
    out[(size_t)b * N_ + tid] = a;
  }
}

// ---------------------------------------------------------------------------
extern "C" void kernel_launch(void* const* d_in, const int* in_sizes, int n_in,
                              void* d_out, int out_size, void* d_ws, size_t ws_size,
                              hipStream_t stream) {
  const int*   ln     = (const int*)d_in[0];
  const int*   rn     = (const int*)d_in[2];
  const float* tlWih  = (const float*)d_in[4];
  const float* tlWhh  = (const float*)d_in[5];
  const float* tlB    = (const float*)d_in[6];
  const float* trWih  = (const float*)d_in[7];
  const float* trWhh  = (const float*)d_in[8];
  const float* trB    = (const float*)d_in[9];
  const float* nlWih0 = (const float*)d_in[10];
  const float* nlWhh0 = (const float*)d_in[11];
  const float* nlB0   = (const float*)d_in[12];
  const float* nlWih1 = (const float*)d_in[13];
  const float* nlWhh1 = (const float*)d_in[14];
  const float* nlB1   = (const float*)d_in[15];
  const float* nrWih0 = (const float*)d_in[16];
  const float* nrWhh0 = (const float*)d_in[17];
  const float* nrB0   = (const float*)d_in[18];
  const float* nrWih1 = (const float*)d_in[19];
  const float* nrWhh1 = (const float*)d_in[20];
  const float* nrB1   = (const float*)d_in[21];
  const float* mW1    = (const float*)d_in[22];
  const float* mb1    = (const float*)d_in[23];
  const float* mW2    = (const float*)d_in[24];
  const float* mb2    = (const float*)d_in[25];

  // workspace layout
  char*     ws     = (char*)d_ws;
  float*    states = (float*)ws;                                 // 2*B*N f32 = 256 KB
  __bf16*   packW  = (__bf16*)(ws + 262144);                     // 6 * 262144 bf16 = 3 MB
  __bf16*   hbase  = (__bf16*)(ws + 262144 + 3145728);           // 4 * 131072 bf16 = 1 MB
  unsigned* ctr    = (unsigned*)(ws + 262144 + 3145728 + 1048576);
  __bf16* h0p0 = hbase;
  __bf16* h0p1 = hbase + 131072;
  __bf16* h1p0 = hbase + 262144;
  __bf16* h1p1 = hbase + 393216;

  init_ws_kernel<<<2048, 256, 0, stream>>>(hbase, ctr);

  pack_weight_kernel<<<1024, 256, 0, stream>>>(nlWhh0, packW + 0 * 262144);
  pack_weight_kernel<<<1024, 256, 0, stream>>>(nlWih1, packW + 1 * 262144);
  pack_weight_kernel<<<1024, 256, 0, stream>>>(nlWhh1, packW + 2 * 262144);
  pack_weight_kernel<<<1024, 256, 0, stream>>>(nrWhh0, packW + 3 * 262144);
  pack_weight_kernel<<<1024, 256, 0, stream>>>(nrWih1, packW + 4 * 262144);
  pack_weight_kernel<<<1024, 256, 0, stream>>>(nrWhh1, packW + 5 * 262144);

  scalar_lstm_kernel<<<dim3(B_, 2), 128, 0, stream>>>(ln, rn, tlWih, tlWhh, tlB,
                                                      trWih, trWhh, trB, states);

  notes_lstm_kernel<<<64, 256, 0, stream>>>(states, packW,
                                            nlWih0, nlB0, nlB1,
                                            nrWih0, nrB0, nrB1,
                                            h0p0, h0p1, h1p0, h1p1, ctr);

  mlp_kernel<<<B_, 256, 0, stream>>>(h1p0, mW1, mb1, mW2, mb2, (float*)d_out);
}
